// CrossFocusedLinearAttention_62079457297001
// MI455X (gfx1250) — matrix operand order repaired and checked by reading, not compiled, not run on hardware
//
#include <hip/hip_runtime.h>
#include <hip/hip_bf16.h>
#include <math.h>
#include <stdint.h>

// ---------------------------------------------------------------------------
// CDNA5 (gfx1250) implementation of CrossFocusedLinearAttention.
// wave32; all GEMMs on v_wmma_f32_16x16x32_bf16; big GEMMs stage the shared
// B panel through LDS with GLOBAL_LOAD_ASYNC_TO_LDS_B128 (double-buffered,
// ASYNCcnt-fenced) to cut redundant L2 traffic; f32 VALU epilogues.
// ---------------------------------------------------------------------------

typedef __bf16 bf16;
typedef __attribute__((ext_vector_type(16))) __bf16 v16bf;
typedef __attribute__((ext_vector_type(8)))  __bf16 v8bf;
typedef __attribute__((ext_vector_type(4)))  __bf16 v4bf;
typedef __attribute__((ext_vector_type(8)))  float  v8f;

#define BQ    8
#define NTOK  4096
#define CDIM  512
#define NH    8
#define HD    64
#define BH    64          // BQ*NH
#define MTOT  32768       // BQ*NTOK
#define EPSF  1e-6f

#define KC     64         // K-chunk staged per LDS buffer
#define LDSTR  72         // padded LDS row stride (bank-conflict-free)

// ---- CDNA5 async global->LDS copy (ASYNCcnt-tracked) -----------------------
__device__ inline void async_copy_b128(const bf16* gsrc, unsigned lds_off) {
    asm volatile("global_load_async_to_lds_b128 %0, %1, off"
                 :: "v"(lds_off),
                    "v"((unsigned long long)(uintptr_t)gsrc)
                 : "memory");
}
__device__ inline void wait_async0() {
    asm volatile("s_wait_asynccnt 0x0" ::: "memory");
}

// ---- WMMA fragment loaders -------------------------------------------------
// A-matrix 16x32 bf16 (ISA 7.12.2): lane holds row (lane&15); lane-half 0 has
// K = {k0+0..7, k0+16..23}, lane-half 1 has K = {k0+8..15, k0+24..31}.
__device__ inline v16bf frag_a_bf16(const bf16* __restrict__ rowp, int k0, int half) {
    const v8bf lo = *(const v8bf*)(rowp + k0 + half * 8);
    const v8bf hi = *(const v8bf*)(rowp + k0 + half * 8 + 16);
    v16bf r;
#pragma unroll
    for (int i = 0; i < 8; ++i) { r[i] = lo[i]; r[8 + i] = hi[i]; }
    return r;
}

__device__ inline v16bf frag_a_f32(const float* __restrict__ rowp, int k0, int half) {
    const float* base = rowp + k0 + half * 8;
    const float4 a0 = *(const float4*)(base);
    const float4 a1 = *(const float4*)(base + 4);
    const float4 b0 = *(const float4*)(base + 16);
    const float4 b1 = *(const float4*)(base + 20);
    v16bf r;
    r[0]  = (bf16)a0.x; r[1]  = (bf16)a0.y; r[2]  = (bf16)a0.z; r[3]  = (bf16)a0.w;
    r[4]  = (bf16)a1.x; r[5]  = (bf16)a1.y; r[6]  = (bf16)a1.z; r[7]  = (bf16)a1.w;
    r[8]  = (bf16)b0.x; r[9]  = (bf16)b0.y; r[10] = (bf16)b0.z; r[11] = (bf16)b0.w;
    r[12] = (bf16)b1.x; r[13] = (bf16)b1.y; r[14] = (bf16)b1.z; r[15] = (bf16)b1.w;
    return r;
}

// B-matrix 32x16 bf16: lane holds column (lane&15); lane-half 0 has
// K = {k0+0..7, k0+16..23}, lane-half 1 has K = {k0+8..15, k0+24..31}.  Source row stored contiguous in K.
__device__ inline v16bf frag_b_bf16(const bf16* rowp, int k0, int half) {
    const v8bf lo = *(const v8bf*)(rowp + k0 + half * 8);
    const v8bf hi = *(const v8bf*)(rowp + k0 + half * 8 + 16);
    v16bf r;
#pragma unroll
    for (int i = 0; i < 8; ++i) { r[i] = lo[i]; r[8 + i] = hi[i]; }
    return r;
}

__device__ inline v8f wmma_bf16(v16bf a, v16bf b, v8f c) {
    return __builtin_amdgcn_wmma_f32_16x16x32_bf16(false, a, false, b, (short)0, c,
                                                   false, false);
}

// ---- small prep kernels ----------------------------------------------------
__global__ void k_wtrans(const float* __restrict__ W, bf16* __restrict__ WT) {
    const int idx = blockIdx.x * blockDim.x + threadIdx.x;  // 0..262143
    const int k = idx >> 9, n = idx & 511;
    WT[(size_t)n * CDIM + k] = (bf16)W[idx];
}

__global__ void k_softplus(const float* __restrict__ s, float* __restrict__ sp) {
    const int c = blockIdx.x * blockDim.x + threadIdx.x;
    if (c < CDIM) {
        const float x = s[c];
        sp[c] = (x > 20.f) ? x : log1pf(expf(x));
    }
}

// ---- big GEMM: M=32768, K=N=512 --------------------------------------------
// 128-thread blocks = 4 waves in a 2x2 grid; each wave owns a 64x64 register
// tile (4x4 v8f accumulators, 16 WMMA per 32-wide k-step).  The 128x512 B
// panel is streamed through double-buffered LDS via async copies; A comes
// straight from global (f32 converted in-register for the input projections).
// MODE 0: relu(x)+eps -> bf16 row-major [M][C]     (q/k projections)
// MODE 1: transposed head-split bf16 [bh][cc][n]   (v projection)
// MODE 2: x + bias    -> f32 row-major [M][C]      (output projection)
template <int AISF32, int MODE>
__global__ __launch_bounds__(128)
void k_gemm_big(const void* __restrict__ Aptr,
                const bf16* __restrict__ Bt,
                void* __restrict__ Out, const float* __restrict__ bias) {
    __shared__ bf16 Bs[2][128][LDSTR];

    const int tid  = threadIdx.x;
    const int lane = tid & 31;
    const int wave = tid >> 5;
    const int half = lane >> 4;
    const int l15  = lane & 15;
    const int wm   = wave >> 1, wn = wave & 1;
    const int m0   = blockIdx.y * 128 + wm * 64;
    const int nblk = blockIdx.x * 128;
    const int n0   = nblk + wn * 64;
    const int K    = CDIM;

    // async-stage one KC-wide slice of the B panel into Bs[bb]
    auto stage = [&](int bb, int kbase) {
#pragma unroll
        for (int i = 0; i < 8; ++i) {
            const int c   = i * 128 + tid;     // 1024 x 16B chunks
            const int row = c >> 3, j = (c & 7) * 8;
            const bf16* g = Bt + (size_t)(nblk + row) * CDIM + kbase + j;
            async_copy_b128(g, (unsigned)(uintptr_t)&Bs[bb][row][j]);
        }
    };

    const float* arow32[4];
    const bf16*  arowbf[4];
#pragma unroll
    for (int t = 0; t < 4; ++t) {
        arow32[t] = (const float*)Aptr + (size_t)(m0 + t * 16 + l15) * K;
        arowbf[t] = (const bf16*)Aptr + (size_t)(m0 + t * 16 + l15) * K;
        __builtin_prefetch((const void*)arowbf[t], 0, 3);   // global_prefetch_b8
    }

    stage(0, 0);
    v8f acc[4][4] = {};
    wait_async0();
    __syncthreads();

    for (int ks = 0; ks < K / KC; ++ks) {
        if (ks + 1 < K / KC) stage((ks + 1) & 1, (ks + 1) * KC);
        const int bb = ks & 1;
#pragma unroll
        for (int kk = 0; kk < KC; kk += 32) {
            const int k0 = ks * KC + kk;
            v16bf a[4];
#pragma unroll
            for (int mi = 0; mi < 4; ++mi)
                a[mi] = AISF32 ? frag_a_f32(arow32[mi], k0, half)
                               : frag_a_bf16(arowbf[mi], k0, half);
#pragma unroll
            for (int ni = 0; ni < 4; ++ni) {
                const v16bf b = frag_b_bf16(&Bs[bb][wn * 64 + ni * 16 + l15][0],
                                            kk, half);
#pragma unroll
                for (int mi = 0; mi < 4; ++mi)
                    acc[mi][ni] = wmma_bf16(a[mi], b, acc[mi][ni]);
            }
        }
        wait_async0();
        __syncthreads();
    }

#pragma unroll
    for (int mi = 0; mi < 4; ++mi) {
#pragma unroll
        for (int ni = 0; ni < 4; ++ni) {
            const int col = n0 + ni * 16 + l15;
#pragma unroll
            for (int r = 0; r < 8; ++r) {
                const int row = m0 + mi * 16 + r + half * 8;
                float v = acc[mi][ni][r];
                if (MODE == 0) {
                    v = fmaxf(v, 0.f) + EPSF;
                    ((bf16*)Out)[(size_t)row * CDIM + col] = (bf16)v;
                } else if (MODE == 1) {
                    const int hh = col >> 6, cc = col & 63;
                    const int b  = row >> 12, n = row & 4095;
                    ((bf16*)Out)[((size_t)((b * NH + hh) * HD + cc)) * NTOK + n] =
                        (bf16)v;
                } else {
                    ((float*)Out)[(size_t)row * CDIM + col] = v + bias[col];
                }
            }
        }
    }
}

// ---- focusing: x/s, x^3, renorm to original L2 per token -------------------
// one wave per token (512 channels). transposed=0 -> qf[bh][n][cc];
// transposed=1 -> kfT[bh][cc][n].
__global__ __launch_bounds__(256)
void k_focus(const bf16* __restrict__ P, const float* __restrict__ sp,
             bf16* __restrict__ Oq, bf16* __restrict__ OkT, int transposed) {
    const int wid  = (blockIdx.x * blockDim.x + threadIdx.x) >> 5;
    const int lane = threadIdx.x & 31;
    const int b = wid >> 12, n = wid & 4095;
    const bf16* row = P + (size_t)wid * CDIM;

    float t3[16];
    float sn = 0.f, st = 0.f;
#pragma unroll
    for (int j = 0; j < 4; ++j) {
        const int c = j * 128 + lane * 4;
        const v4bf pv = *(const v4bf*)(row + c);
#pragma unroll
        for (int l = 0; l < 4; ++l) {
            const float v = (float)pv[l] / sp[c + l];
            const float t = v * v * v;
            sn += v * v;
            st += t * t;
            t3[j * 4 + l] = t;
        }
    }
#pragma unroll
    for (int o = 16; o >= 1; o >>= 1) {
        sn += __shfl_xor(sn, o, 32);
        st += __shfl_xor(st, o, 32);
    }
    const float ratio = sqrtf(sn) * rsqrtf(st + 1e-38f);

#pragma unroll
    for (int j = 0; j < 4; ++j) {
        const int c  = j * 128 + lane * 4;
        const int hh = c >> 6, cc = c & 63;
        if (!transposed) {
            v4bf o;
#pragma unroll
            for (int l = 0; l < 4; ++l) o[l] = (bf16)(t3[j * 4 + l] * ratio);
            *(v4bf*)(Oq + ((size_t)((b * NH + hh) * NTOK + n)) * HD + cc) = o;
        } else {
#pragma unroll
            for (int l = 0; l < 4; ++l)
                OkT[((size_t)((b * NH + hh) * HD + cc + l)) * NTOK + n] =
                    (bf16)(t3[j * 4 + l] * ratio);
        }
    }
}

// ---- ksum[bh][cc] = sum_n kfT[bh][cc][n] (one wave per row of 4096) --------
__global__ __launch_bounds__(256)
void k_ksum(const bf16* __restrict__ KT, float* __restrict__ KS) {
    const int rid  = (blockIdx.x * blockDim.x + threadIdx.x) >> 5;  // 0..4095
    const int lane = threadIdx.x & 31;
    const bf16* row = KT + (size_t)rid * NTOK;
    float s = 0.f;
    for (int it = 0; it < 16; ++it) {
        const v8bf v = *(const v8bf*)(row + it * 256 + lane * 8);
#pragma unroll
        for (int i = 0; i < 8; ++i) s += (float)v[i];
    }
#pragma unroll
    for (int o = 16; o >= 1; o >>= 1) s += __shfl_xor(s, o, 32);
    if (lane == 0) KS[rid] = s;
}

// ---- z[bh][n] = 1 / (qf[bh][n][:] . ksum[bh][:] + eps) ---------------------
__global__ __launch_bounds__(256)
void k_z(const bf16* __restrict__ QF, const float* __restrict__ KS,
         float* __restrict__ Z) {
    const int tid = blockIdx.x * blockDim.x + threadIdx.x;  // 0..262143
    const int bh = tid >> 12;
    const bf16* q = QF + (size_t)tid * HD;
    const float* ks = KS + bh * HD;
    float d = 0.f;
#pragma unroll 8
    for (int i = 0; i < 64; ++i) d += (float)q[i] * ks[i];
    Z[tid] = 1.f / (d + EPSF);
}

// ---- batched kvT[bh][d][c] = sum_j v[j][d] k[j][c]  (M=N=64, K=4096) -------
// one wave per 32x32 tile: 4 WMMA : 4 frag loads per k-step.
__global__ __launch_bounds__(32)
void k_gemm_kv(const bf16* __restrict__ VT, const bf16* __restrict__ KT,
               bf16* __restrict__ KV) {
    const int lane = threadIdx.x & 31;
    const int half = lane >> 4, l15 = lane & 15;
    const int bh = blockIdx.z;
    const int m0 = blockIdx.y * 32;  // d
    const int n0 = blockIdx.x * 32;  // c
    const bf16* arow[2];
    const bf16* brow[2];
#pragma unroll
    for (int t = 0; t < 2; ++t) {
        arow[t] = VT + ((size_t)bh * HD + m0 + t * 16 + l15) * NTOK;
        brow[t] = KT + ((size_t)bh * HD + n0 + t * 16 + l15) * NTOK;
        __builtin_prefetch((const void*)arow[t], 0, 3);
        __builtin_prefetch((const void*)brow[t], 0, 3);
    }
    v8f acc[2][2] = {};
    for (int k0 = 0; k0 < NTOK; k0 += 32) {
        v16bf a[2];
#pragma unroll
        for (int mi = 0; mi < 2; ++mi) a[mi] = frag_a_bf16(arow[mi], k0, half);
#pragma unroll
        for (int ni = 0; ni < 2; ++ni) {
            const v16bf b = frag_b_bf16(brow[ni], k0, half);
#pragma unroll
            for (int mi = 0; mi < 2; ++mi)
                acc[mi][ni] = wmma_bf16(a[mi], b, acc[mi][ni]);
        }
    }
#pragma unroll
    for (int mi = 0; mi < 2; ++mi) {
#pragma unroll
        for (int ni = 0; ni < 2; ++ni) {
#pragma unroll
            for (int r = 0; r < 8; ++r) {
                const int d = m0 + mi * 16 + r + half * 8;
                const int c = n0 + ni * 16 + l15;
                KV[((size_t)bh * HD + d) * HD + c] = (bf16)acc[mi][ni][r];
            }
        }
    }
}

// ---- batched x[bh][n][d] = z[bh][n] * (qf[bh][n][:] @ kv[:][d]) ------------
// one wave per 16(row)x64(col) tile; K=64 (2 k-steps, A reused over 4 B).
// stored merged-head bf16: xf[b][n][h*64+d]
__global__ __launch_bounds__(32)
void k_gemm_x(const bf16* __restrict__ QF, const bf16* __restrict__ KV,
              const float* __restrict__ Z, bf16* __restrict__ XF) {
    const int lane = threadIdx.x & 31;
    const int half = lane >> 4, l15 = lane & 15;
    const int bh  = blockIdx.y;
    const int n0t = blockIdx.x * 16;  // token rows
    const bf16* arow = QF + ((size_t)bh * NTOK + n0t + l15) * HD;
    const bf16* brow[4];
#pragma unroll
    for (int t = 0; t < 4; ++t)
        brow[t] = KV + ((size_t)bh * HD + t * 16 + l15) * HD;

    v8f acc[4] = {};
#pragma unroll
    for (int k0 = 0; k0 < HD; k0 += 32) {
        const v16bf a = frag_a_bf16(arow, k0, half);
#pragma unroll
        for (int t = 0; t < 4; ++t) {
            const v16bf b = frag_b_bf16(brow[t], k0, half);
            acc[t] = wmma_bf16(a, b, acc[t]);
        }
    }
    const int b = bh >> 3, hh = bh & 7;
#pragma unroll
    for (int t = 0; t < 4; ++t) {
#pragma unroll
        for (int r = 0; r < 8; ++r) {
            const int n = n0t + r + half * 8, d = t * 16 + l15;
            const float v = acc[t][r] * Z[(size_t)bh * NTOK + n];
            XF[((size_t)(b * NTOK + n)) * CDIM + hh * HD + d] = (bf16)v;
        }
    }
}

// ---------------------------------------------------------------------------
extern "C" void kernel_launch(void* const* d_in, const int* in_sizes, int n_in,
                              void* d_out, int out_size, void* d_ws, size_t ws_size,
                              hipStream_t stream) {
    (void)in_sizes; (void)n_in; (void)out_size; (void)ws_size;
    const float* query  = (const float*)d_in[0];
    const float* key_in = (const float*)d_in[1];
    const float* value  = (const float*)d_in[2];
    const float* Wq     = (const float*)d_in[3];
    const float* Wk     = (const float*)d_in[4];
    const float* Wv     = (const float*)d_in[5];
    const float* Wp     = (const float*)d_in[6];
    const float* bp     = (const float*)d_in[7];
    const float* scale  = (const float*)d_in[8];
    // d_in[9], d_in[10]: H, W (unused)

    char* p = (char*)d_ws;
    auto carve = [&](size_t bytes) -> void* {
        void* r = (void*)p;
        p += (bytes + 255) & ~(size_t)255;
        return r;
    };
    bf16*  WqT  = (bf16*)carve((size_t)CDIM * CDIM * sizeof(bf16));
    bf16*  WkT  = (bf16*)carve((size_t)CDIM * CDIM * sizeof(bf16));
    bf16*  WvT  = (bf16*)carve((size_t)CDIM * CDIM * sizeof(bf16));
    bf16*  WpT  = (bf16*)carve((size_t)CDIM * CDIM * sizeof(bf16));
    float* sp   = (float*)carve((size_t)CDIM * sizeof(float));
    bf16*  qp   = (bf16*)carve((size_t)MTOT * CDIM * sizeof(bf16));
    bf16*  kp   = (bf16*)carve((size_t)MTOT * CDIM * sizeof(bf16));
    bf16*  vfT  = (bf16*)carve((size_t)BH * HD * NTOK * sizeof(bf16));
    bf16*  qf   = (bf16*)carve((size_t)BH * NTOK * HD * sizeof(bf16));
    bf16*  kfT  = (bf16*)carve((size_t)BH * HD * NTOK * sizeof(bf16));
    float* ksum = (float*)carve((size_t)BH * HD * sizeof(float));
    float* z    = (float*)carve((size_t)BH * NTOK * sizeof(float));
    bf16*  kvT  = (bf16*)carve((size_t)BH * HD * HD * sizeof(bf16));
    bf16*  xf   = (bf16*)carve((size_t)MTOT * CDIM * sizeof(bf16));

    // weight transposes (f32 -> bf16 [N][K]) + softplus(scale)
    k_wtrans<<<1024, 256, 0, stream>>>(Wq, WqT);
    k_wtrans<<<1024, 256, 0, stream>>>(Wk, WkT);
    k_wtrans<<<1024, 256, 0, stream>>>(Wv, WvT);
    k_wtrans<<<1024, 256, 0, stream>>>(Wp, WpT);
    k_softplus<<<2, 256, 0, stream>>>(scale, sp);

    // projections via WMMA (f32 A converted in-register, B async-staged in LDS)
    const dim3 gg(CDIM / 128, MTOT / 128);  // (4, 256)
    k_gemm_big<1, 0><<<gg, 128, 0, stream>>>(query,  WqT, qp,  nullptr);
    k_gemm_big<1, 0><<<gg, 128, 0, stream>>>(key_in, WkT, kp,  nullptr);
    k_gemm_big<1, 1><<<gg, 128, 0, stream>>>(value,  WvT, vfT, nullptr);

    // focusing (scale divide, x^3 renorm) + head split / transpose
    k_focus<<<MTOT * 32 / 256, 256, 0, stream>>>(qp, sp, qf, nullptr, 0);
    k_focus<<<MTOT * 32 / 256, 256, 0, stream>>>(kp, sp, nullptr, kfT, 1);

    // z-normalizer
    k_ksum<<<BH * HD * 32 / 256, 256, 0, stream>>>(kfT, ksum);
    k_z<<<BH * NTOK / 256, 256, 0, stream>>>(qf, ksum, z);

    // kv aggregation and q @ kv
    k_gemm_kv<<<dim3(2, 2, BH), 32, 0, stream>>>(vfT, kfT, kvT);
    k_gemm_x<<<dim3(NTOK / 16, BH), 32, 0, stream>>>(qf, kvT, z, xf);

    // output projection + bias -> f32 d_out
    k_gemm_big<0, 2><<<gg, 128, 0, stream>>>(xf, WpT, d_out, bp);
}
